// JambaMambaDecoderLayer_72902774882378
// MI455X (gfx1250) — matrix operand order repaired and checked by reading, not compiled, not run on hardware
//
#include <hip/hip_runtime.h>
#include <math.h>

// Problem constants (match reference)
#define L_SEQ 2048
#define H_DIM 1024
#define DMODEL 2048
#define NSTATE 16
#define RRANK 64
#define NEXP 8
#define IDIM 2048

typedef __bf16 bf16;
typedef bf16 bf16x8  __attribute__((ext_vector_type(8)));
typedef bf16 bf16x16 __attribute__((ext_vector_type(16)));
typedef float v8f    __attribute__((ext_vector_type(8)));
typedef int i32x4 __attribute__((vector_size(16)));

__device__ __forceinline__ float sigmoidf_(float x) { return 1.f / (1.f + __expf(-x)); }

// ---------------------------------------------------------------------------
// CDNA5 async global->LDS copy (ASYNCcnt) with synchronous fallback
// ---------------------------------------------------------------------------
#if defined(__has_builtin)
#if __has_builtin(__builtin_amdgcn_global_load_async_to_lds_b128)
#define HAVE_ASYNC 1
#endif
#endif
#ifndef HAVE_ASYNC
#define HAVE_ASYNC 0
#endif

__device__ __forceinline__ void async_copy_b128(const void* g, void* l) {
#if HAVE_ASYNC
  __builtin_amdgcn_global_load_async_to_lds_b128(
      (i32x4*)g, (__attribute__((address_space(3))) i32x4*)l, 0, 0);
#else
  *(bf16x8*)l = *(const bf16x8*)g;
#endif
}

#if HAVE_ASYNC
#if __has_builtin(__builtin_amdgcn_s_wait_asynccnt)
#define WAIT_ASYNC(n) __builtin_amdgcn_s_wait_asynccnt(n)
#else
#define WAIT_ASYNC(n) asm volatile("s_wait_asynccnt %0" ::"i"(n) : "memory")
#endif
#else
#define WAIT_ASYNC(n) ((void)0)
#endif

__device__ __forceinline__ bf16x16 load_frag16(const bf16* row, int o1, int o2) {
  bf16x8 lo = *(const bf16x8*)(row + o1);
  bf16x8 hi = *(const bf16x8*)(row + o2);
  bf16x16 r;
#pragma unroll
  for (int j = 0; j < 8; ++j) {
    r[j] = lo[j];
    r[j + 8] = hi[j];
  }
  return r;
}

__device__ __forceinline__ v8f wmma_bf16(bf16x16 a, bf16x16 b, v8f c) {
  return __builtin_amdgcn_wmma_f32_16x16x32_bf16(false, a, false, b, (short)0, c,
                                                 false, false);
}

// ---------------------------------------------------------------------------
// WMMA GEMM (all inputs bf16, weights pre-transposed to [N,K]):
//   C[M,N] = epilogue( A[M,K] * B[N,K]^T  [, A * B2^T] )
//   GROUPED: blockIdx.z = expert e; Meff=counts[e]; rows offset offs[e];
//            B/B2 += e*bStride
//   AIND   : A row fetched through row_map (token gather)
//   DUAL   : second weight matrix B2 (fused gate+up)
//   EPI: 0 = store f32 C[srow,n]
//        1 = store f32 softplus(acc + aux[n])              (dt projection)
//        2 = store f32 acc + aux[m,ldaux]                  (residual add)
//        3 = atomicAdd f32 C[token,n] += w_map*acc         (MoE combine)
//        5 = store bf16 silu(accA)*accB                    (fused act)
//        6 = store bf16 C[srow,n]                          (in_proj -> xz)
// Tiles: BM=64, BN=64, BK=32; 8 waves -> each wave 16(m) x 32(n) = 2 subtiles.
// Double-buffered LDS fed by async-to-LDS copies.
// ---------------------------------------------------------------------------
#define BM 64
#define BN 64
#define BK 32
#define LDK (BK + 8)

template <int EPI, bool GROUPED, bool AIND, bool DUAL>
__global__ __launch_bounds__(256) void gemm_wmma(
    const bf16* __restrict__ A, int lda, const bf16* __restrict__ B,
    const bf16* __restrict__ B2, long long bStride, int ldb, void* __restrict__ Cv,
    int ldc, const float* __restrict__ aux, int ldaux, int M, int N, int K,
    const int* __restrict__ counts, const int* __restrict__ offs,
    const int* __restrict__ row_map, const float* __restrict__ w_map) {
  const int tid = threadIdx.x;
  const int e = blockIdx.z;
  int Meff = M, off = 0;
  if (GROUPED) {
    Meff = counts[e];
    off = offs[e];
    B += (long long)e * bStride;
    if (DUAL) B2 += (long long)e * bStride;
  }
  const int m0 = blockIdx.y * BM;
  const int n0 = blockIdx.x * BN;
  if (m0 >= Meff) return;  // block-uniform: EXEC all-ones at every WMMA

  __shared__ __align__(16) bf16 As[2][BM][LDK];
  __shared__ __align__(16) bf16 Bs[2][BN][LDK];
  __shared__ __align__(16) bf16 B2s[DUAL ? 2 : 1][BN][LDK];

  // --- tile loader mapping: 256 threads x one 16B chunk per matrix ---
  const int lrow = tid >> 2;        // 0..63
  const int lkc = (tid & 3) * 8;    // bf16 elem offset 0,8,16,24
  long long arow;
  {
    int m = m0 + lrow;
    int mm = (m < Meff) ? m : (Meff - 1);
    arow = GROUPED ? (AIND ? (long long)row_map[off + mm] : (long long)(off + mm))
                   : (long long)mm;
  }
  const bf16* Ag = A + arow * lda + lkc;
  int ng = n0 + lrow;
  if (ng >= N) ng = N - 1;  // clamp; columns >= N never stored
  const bf16* Bg = B + (long long)ng * ldb + lkc;
  const bf16* B2g = DUAL ? (B2 + (long long)ng * ldb + lkc) : nullptr;

  // --- wave / fragment mapping ---
  const int lane = tid & 31;
  const int wid = tid >> 5;
  const int wm = wid >> 1;        // 0..3 : m sub-tile
  const int wn2 = (wid & 1) * 2;  // 0/2  : first of two n sub-tiles
  const int khalf = lane >> 4;
  const int l16 = lane & 15;

  v8f accA[2] = {v8f{}, v8f{}};
  v8f accB[2] = {v8f{}, v8f{}};

  const int nk = K / BK;
  // stage 0
  async_copy_b128(Ag, &As[0][lrow][lkc]);
  async_copy_b128(Bg, &Bs[0][lrow][lkc]);
  if constexpr (DUAL) async_copy_b128(B2g, &B2s[0][lrow][lkc]);

  for (int i = 0; i < nk; ++i) {
    const int p = i & 1;
    if (i + 1 < nk) {  // prefetch next stage into other buffer
      const int k1 = (i + 1) * BK;
      async_copy_b128(Ag + k1, &As[p ^ 1][lrow][lkc]);
      async_copy_b128(Bg + k1, &Bs[p ^ 1][lrow][lkc]);
      if constexpr (DUAL) async_copy_b128(B2g + k1, &B2s[p ^ 1][lrow][lkc]);
      if constexpr (DUAL) {
        WAIT_ASYNC(3);
      } else {
        WAIT_ASYNC(2);
      }
    } else {
      WAIT_ASYNC(0);
    }
    __syncthreads();

    bf16x16 af = load_frag16(&As[p][wm * 16 + l16][0], khalf * 16, khalf * 16 + 8);
#pragma unroll
    for (int s = 0; s < 2; ++s) {
      bf16x16 bf0 =
          load_frag16(&Bs[p][(wn2 + s) * 16 + l16][0], khalf * 16, khalf * 16 + 8);
      accA[s] = wmma_bf16(af, bf0, accA[s]);
      if constexpr (DUAL) {
        bf16x16 bf1 =
            load_frag16(&B2s[p][(wn2 + s) * 16 + l16][0], khalf * 16, khalf * 16 + 8);
        accB[s] = wmma_bf16(af, bf1, accB[s]);
      }
    }
    __syncthreads();
  }

  // --- epilogue; C layout: m = base + r + 8*khalf, n = l16 ---
  float* Cf = (float*)Cv;
  bf16* Cb = (bf16*)Cv;
  const int mB = m0 + wm * 16 + 8 * khalf;
#pragma unroll
  for (int s = 0; s < 2; ++s) {
    const int cn = n0 + (wn2 + s) * 16 + l16;
    if (cn >= N) continue;
#pragma unroll
    for (int r = 0; r < 8; ++r) {
      const int m = mB + r;
      if (m >= Meff) continue;
      float v = accA[s][r];
      if constexpr (EPI == 0) {
        long long srow = GROUPED ? (long long)(off + m) : (long long)m;
        Cf[srow * ldc + cn] = v;
      } else if constexpr (EPI == 1) {
        v += aux[cn];
        v = (v > 20.f) ? v : log1pf(__expf(v));
        Cf[(long long)m * ldc + cn] = v;
      } else if constexpr (EPI == 2) {
        v += aux[(long long)m * ldaux + cn];
        Cf[(long long)m * ldc + cn] = v;
      } else if constexpr (EPI == 3) {
        int srow = off + m;
        atomicAdd(&Cf[(long long)row_map[srow] * ldc + cn], w_map[srow] * v);
      } else if constexpr (EPI == 5) {
        long long srow = off + m;
        Cb[srow * ldc + cn] = (bf16)(v * sigmoidf_(v) * accB[s][r]);
      } else if constexpr (EPI == 6) {
        Cb[(long long)m * ldc + cn] = (bf16)v;
      }
    }
  }
}

// ---------------------------------------------------------------------------
// Weight preparation
// ---------------------------------------------------------------------------
__global__ __launch_bounds__(256) void cvt_bf16_kernel(const float* __restrict__ in,
                                                       bf16* __restrict__ out,
                                                       long long n) {
  long long i = (long long)blockIdx.x * 256 + threadIdx.x;
  if (i < n) out[i] = (bf16)in[i];
}

// in: [E][K][N] f32  ->  out: [E][N][K] bf16  (tiled transpose)
__global__ __launch_bounds__(256) void transpose_cvt_kernel(const float* __restrict__ in,
                                                            bf16* __restrict__ out,
                                                            int K, int N) {
  __shared__ float tile[32][33];
  const int e = blockIdx.z;
  const float* src = in + (long long)e * K * N;
  bf16* dst = out + (long long)e * K * N;
  const int n0 = blockIdx.x * 32, k0 = blockIdx.y * 32;
  const int tx = threadIdx.x & 31, ty = threadIdx.x >> 5;
#pragma unroll
  for (int i = 0; i < 32; i += 8)
    tile[ty + i][tx] = src[(long long)(k0 + ty + i) * N + (n0 + tx)];
  __syncthreads();
#pragma unroll
  for (int i = 0; i < 32; i += 8)
    dst[(long long)(n0 + ty + i) * K + (k0 + tx)] = (bf16)tile[tx][ty + i];
}

// ---------------------------------------------------------------------------
// Elementwise / reduction kernels
// ---------------------------------------------------------------------------
__global__ __launch_bounds__(256) void rmsnorm_kernel(const float* __restrict__ in,
                                                      const float* __restrict__ w,
                                                      float* __restrict__ outf,
                                                      bf16* __restrict__ outb, int n) {
  int t = blockIdx.x;
  const float* x = in + (long long)t * n;
  float s = 0.f;
  for (int i = threadIdx.x; i < n; i += 256) {
    float v = x[i];
    s += v * v;
  }
  __shared__ float red[256];
  red[threadIdx.x] = s;
  __syncthreads();
  for (int o = 128; o > 0; o >>= 1) {
    if (threadIdx.x < o) red[threadIdx.x] += red[threadIdx.x + o];
    __syncthreads();
  }
  float inv = rsqrtf(red[0] / n + 1e-6f);
  for (int i = threadIdx.x; i < n; i += 256) {
    float v = x[i] * inv * w[i];
    outf[(long long)t * n + i] = v;
    outb[(long long)t * n + i] = (bf16)v;
  }
}

__global__ __launch_bounds__(256) void conv_silu_kernel(const bf16* __restrict__ xz,
                                                        const float* __restrict__ cw,
                                                        const float* __restrict__ cb,
                                                        float* __restrict__ xc,
                                                        bf16* __restrict__ xcb) {
  int d = blockIdx.x * 256 + threadIdx.x;
  int t = blockIdx.y;
  float acc = cb[d];
#pragma unroll
  for (int k = 0; k < 4; ++k) {
    int tt = t - 3 + k;
    if (tt >= 0) acc += (float)xz[(long long)tt * (2 * DMODEL) + d] * cw[d * 4 + k];
  }
  float v = acc * sigmoidf_(acc);
  xc[(long long)t * DMODEL + d] = v;
  xcb[(long long)t * DMODEL + d] = (bf16)v;
}

__global__ __launch_bounds__(96) void small_norms_kernel(
    const float* __restrict__ proj, const float* __restrict__ dtw,
    const float* __restrict__ bw, const float* __restrict__ cw,
    bf16* __restrict__ dtnb, float* __restrict__ Bn, float* __restrict__ Cn) {
  int t = blockIdx.x;
  int tid = threadIdx.x;
  __shared__ float s[96];
  __shared__ float inv[3];
  s[tid] = proj[(long long)t * 96 + tid];
  __syncthreads();
  if (tid < 3) {
    int base = (tid == 0) ? 0 : (tid == 1 ? 64 : 80);
    int len = (tid == 0) ? 64 : 16;
    float acc = 0.f;
    for (int i = 0; i < len; ++i) {
      float v = s[base + i];
      acc += v * v;
    }
    inv[tid] = rsqrtf(acc / len + 1e-6f);
  }
  __syncthreads();
  if (tid < 64)
    dtnb[(long long)t * 64 + tid] = (bf16)(s[tid] * inv[0] * dtw[tid]);
  else if (tid < 80)
    Bn[(long long)t * 16 + (tid - 64)] = s[tid] * inv[1] * bw[tid - 64];
  else
    Cn[(long long)t * 16 + (tid - 80)] = s[tid] * inv[2] * cw[tid - 80];
}

// Selective scan: lane = state n (16 per half-wave), 2 channels per wave.
__global__ __launch_bounds__(256) void scan_kernel(
    const float* __restrict__ dt, const float* __restrict__ xc,
    const float* __restrict__ Bn, const float* __restrict__ Cn,
    const float* __restrict__ A_log, float* __restrict__ ys) {
  int lane = threadIdx.x & 31;
  int wid = threadIdx.x >> 5;
  int n = lane & 15;
  int d = (blockIdx.x * 8 + wid) * 2 + (lane >> 4);
  float a_c = -__expf(A_log[d * NSTATE + n]);
  float h = 0.f;
  for (int t = 0; t < L_SEQ; ++t) {
    float dtv = dt[(long long)t * DMODEL + d];
    float xv = xc[(long long)t * DMODEL + d];
    float bv = Bn[t * NSTATE + n];
    float cv = Cn[t * NSTATE + n];
    float dA = __expf(dtv * a_c);
    h = h * dA + dtv * xv * bv;
    float p = h * cv;
#pragma unroll
    for (int o = 1; o < 16; o <<= 1) p += __shfl_xor(p, o, 32);
    if (n == 0) ys[(long long)t * DMODEL + d] = p;
  }
}

__global__ __launch_bounds__(256) void gate_kernel(const bf16* __restrict__ xz,
                                                   const float* __restrict__ xc,
                                                   const float* __restrict__ D,
                                                   const float* __restrict__ ys,
                                                   bf16* __restrict__ ysb) {
  long long i = (long long)blockIdx.x * 256 + threadIdx.x;
  int t = (int)(i / DMODEL);
  int d = (int)(i % DMODEL);
  float y = ys[i] + xc[i] * D[d];
  float z = (float)xz[(long long)t * (2 * DMODEL) + DMODEL + d];
  ysb[i] = (bf16)(y * (z * sigmoidf_(z)));
}

__global__ __launch_bounds__(256) void router_kernel(const float* __restrict__ x2,
                                                     const float* __restrict__ rw,
                                                     int* __restrict__ top_idx,
                                                     float* __restrict__ top_w) {
  int t = blockIdx.x;
  int e = threadIdx.x >> 5;
  int lane = threadIdx.x & 31;
  const float* x = x2 + (long long)t * H_DIM;
  const float* w = rw + (long long)e * H_DIM;
  float s = 0.f;
  for (int i = lane; i < H_DIM; i += 32) s += x[i] * w[i];
#pragma unroll
  for (int o = 16; o > 0; o >>= 1) s += __shfl_down(s, o, 32);
  __shared__ float lg[NEXP];
  if (lane == 0) lg[e] = s;
  __syncthreads();
  if (threadIdx.x == 0) {
    float mx = lg[0];
    for (int i = 1; i < NEXP; ++i) mx = fmaxf(mx, lg[i]);
    float pe[NEXP];
    float sum = 0.f;
    for (int i = 0; i < NEXP; ++i) {
      pe[i] = __expf(lg[i] - mx);
      sum += pe[i];
    }
    for (int i = 0; i < NEXP; ++i) pe[i] /= sum;
    int i0 = 0;
    for (int i = 1; i < NEXP; ++i)
      if (pe[i] > pe[i0]) i0 = i;
    int i1 = (i0 == 0) ? 1 : 0;
    for (int i = 0; i < NEXP; ++i) {
      if (i == i0) continue;
      if (pe[i] > pe[i1]) i1 = i;
    }
    top_idx[t * 2] = i0;
    top_idx[t * 2 + 1] = i1;
    top_w[t * 2] = pe[i0];
    top_w[t * 2 + 1] = pe[i1];
  }
}

__global__ __launch_bounds__(256) void count_kernel(const int* __restrict__ top_idx,
                                                    int* __restrict__ counts) {
  int t = blockIdx.x * 256 + threadIdx.x;
  if (t >= L_SEQ) return;
  atomicAdd(&counts[top_idx[t * 2]], 1);
  atomicAdd(&counts[top_idx[t * 2 + 1]], 1);
}

__global__ void offsets_kernel(const int* __restrict__ counts, int* __restrict__ offs) {
  if (threadIdx.x == 0 && blockIdx.x == 0) {
    int o = 0;
    for (int e = 0; e < NEXP; ++e) {
      offs[e] = o;
      o += counts[e];
    }
  }
}

__global__ __launch_bounds__(256) void scatter_kernel(
    const int* __restrict__ top_idx, const float* __restrict__ top_w,
    const int* __restrict__ offs, int* __restrict__ cursors,
    int* __restrict__ rows_token, float* __restrict__ rows_weight) {
  int t = blockIdx.x * 256 + threadIdx.x;
  if (t >= L_SEQ) return;
#pragma unroll
  for (int j = 0; j < 2; ++j) {
    int e = top_idx[t * 2 + j];
    int pos = atomicAdd(&cursors[e], 1);
    int r = offs[e] + pos;
    rows_token[r] = t;
    rows_weight[r] = top_w[t * 2 + j];
  }
}

// ---------------------------------------------------------------------------
extern "C" void kernel_launch(void* const* d_in, const int* in_sizes, int n_in,
                              void* d_out, int out_size, void* d_ws, size_t ws_size,
                              hipStream_t stream) {
  const float* hidden = (const float*)d_in[0];
  const float* input_ln_w = (const float*)d_in[1];
  const float* in_proj_w = (const float*)d_in[2];
  const float* conv_w = (const float*)d_in[3];
  const float* conv_b = (const float*)d_in[4];
  const float* x_proj_w = (const float*)d_in[5];
  const float* dt_norm_w = (const float*)d_in[6];
  const float* b_norm_w = (const float*)d_in[7];
  const float* c_norm_w = (const float*)d_in[8];
  const float* dt_proj_w = (const float*)d_in[9];
  const float* dt_proj_b = (const float*)d_in[10];
  const float* A_log = (const float*)d_in[11];
  const float* D_param = (const float*)d_in[12];
  const float* out_proj_w = (const float*)d_in[13];
  const float* pre_ff_ln_w = (const float*)d_in[14];
  const float* router_w = (const float*)d_in[15];
  const float* w_gate = (const float*)d_in[16];
  const float* w_up = (const float*)d_in[17];
  const float* w_down = (const float*)d_in[18];

  float* out = (float*)d_out;                                  // [L,H] MoE output
  float* residual = (float*)d_out + (long long)L_SEQ * H_DIM;  // [L,H]

  // ---- workspace layout ----
  float* f = (float*)d_ws;
  float* xf = f;   f += (long long)L_SEQ * H_DIM;  // f32 rmsnorm out (router uses x2)
  float* xc = f;   f += (long long)L_SEQ * DMODEL;
  float* proj = f; f += (long long)L_SEQ * 96;
  float* Bn = f;   f += (long long)L_SEQ * NSTATE;
  float* Cn = f;   f += (long long)L_SEQ * NSTATE;
  float* dt = f;   f += (long long)L_SEQ * DMODEL;
  float* ys = f;   f += (long long)L_SEQ * DMODEL;
  float* top_w = f;       f += L_SEQ * 2;
  float* rows_weight = f; f += 2 * L_SEQ;
  int* ip = (int*)f;
  int* top_idx = ip;    ip += L_SEQ * 2;
  int* counts = ip;     ip += NEXP;
  int* cursors = ip;    ip += NEXP;
  int* offs = ip;       ip += NEXP;
  int* rows_token = ip; ip += 2 * L_SEQ;
  bf16* b = (bf16*)ip;
  bf16* xnb = b;   b += (long long)L_SEQ * H_DIM;      // A of in_proj; later x2 bf16
  bf16* xzb = b;   b += (long long)L_SEQ * 2 * DMODEL;
  bf16* xcb = b;   b += (long long)L_SEQ * DMODEL;
  bf16* dtnb = b;  b += (long long)L_SEQ * RRANK;
  bf16* ysb = b;   b += (long long)L_SEQ * DMODEL;
  bf16* actb = b;  b += (long long)2 * L_SEQ * IDIM;   // gathered silu(g)*u rows
  bf16* wb_in = b;   b += (long long)2 * DMODEL * H_DIM;
  bf16* wb_xp = b;   b += (long long)96 * DMODEL;
  bf16* wb_dt = b;   b += (long long)DMODEL * RRANK;
  bf16* wb_out = b;  b += (long long)H_DIM * DMODEL;
  bf16* wb_gate = b; b += (long long)NEXP * IDIM * H_DIM;  // transposed [E][I][H]
  bf16* wb_up = b;   b += (long long)NEXP * IDIM * H_DIM;  // transposed [E][I][H]
  bf16* wb_down = b; b += (long long)NEXP * H_DIM * IDIM;  // transposed [E][H][I]

  (void)hipMemsetAsync(out, 0, (size_t)L_SEQ * H_DIM * sizeof(float), stream);
  (void)hipMemsetAsync(counts, 0, 2 * NEXP * sizeof(int), stream);  // counts+cursors

  // ---- weight prep: bf16 (+transpose for MoE so every GEMM is x @ W^T) ----
  {
    long long n1 = (long long)2 * DMODEL * H_DIM;
    cvt_bf16_kernel<<<(int)((n1 + 255) / 256), 256, 0, stream>>>(in_proj_w, wb_in, n1);
    long long n2 = (long long)96 * DMODEL;
    cvt_bf16_kernel<<<(int)((n2 + 255) / 256), 256, 0, stream>>>(x_proj_w, wb_xp, n2);
    long long n3 = (long long)DMODEL * RRANK;
    cvt_bf16_kernel<<<(int)((n3 + 255) / 256), 256, 0, stream>>>(dt_proj_w, wb_dt, n3);
    long long n4 = (long long)H_DIM * DMODEL;
    cvt_bf16_kernel<<<(int)((n4 + 255) / 256), 256, 0, stream>>>(out_proj_w, wb_out, n4);
    transpose_cvt_kernel<<<dim3(IDIM / 32, H_DIM / 32, NEXP), 256, 0, stream>>>(
        w_gate, wb_gate, H_DIM, IDIM);
    transpose_cvt_kernel<<<dim3(IDIM / 32, H_DIM / 32, NEXP), 256, 0, stream>>>(
        w_up, wb_up, H_DIM, IDIM);
    transpose_cvt_kernel<<<dim3(H_DIM / 32, IDIM / 32, NEXP), 256, 0, stream>>>(
        w_down, wb_down, IDIM, H_DIM);
  }

  // 1) x = rmsnorm(hidden)
  rmsnorm_kernel<<<L_SEQ, 256, 0, stream>>>(hidden, input_ln_w, xf, xnb, H_DIM);

  // 2) xz = x @ in_proj^T  [2048 x 4096 x 1024] -> bf16
  gemm_wmma<6, false, false, false>
      <<<dim3(4096 / BN, L_SEQ / BM, 1), 256, 0, stream>>>(
          xnb, H_DIM, wb_in, nullptr, 0, H_DIM, xzb, 2 * DMODEL, nullptr, 0, L_SEQ,
          2 * DMODEL, H_DIM, nullptr, nullptr, nullptr, nullptr);

  // 3) causal conv (K=4) + silu
  conv_silu_kernel<<<dim3(DMODEL / 256, L_SEQ), 256, 0, stream>>>(xzb, conv_w, conv_b,
                                                                  xc, xcb);

  // 4) proj = xc @ x_proj^T  [2048 x 96 x 2048]
  gemm_wmma<0, false, false, false><<<dim3(2, L_SEQ / BM, 1), 256, 0, stream>>>(
      xcb, DMODEL, wb_xp, nullptr, 0, DMODEL, proj, 96, nullptr, 0, L_SEQ, 96, DMODEL,
      nullptr, nullptr, nullptr, nullptr);

  // 5) rmsnorm dt_low / B / C
  small_norms_kernel<<<L_SEQ, 96, 0, stream>>>(proj, dt_norm_w, b_norm_w, c_norm_w,
                                               dtnb, Bn, Cn);

  // 6) dt = softplus(dtn @ dt_proj^T + b)  [2048 x 2048 x 64]
  gemm_wmma<1, false, false, false>
      <<<dim3(DMODEL / BN, L_SEQ / BM, 1), 256, 0, stream>>>(
          dtnb, RRANK, wb_dt, nullptr, 0, RRANK, dt, DMODEL, dt_proj_b, 0, L_SEQ,
          DMODEL, RRANK, nullptr, nullptr, nullptr, nullptr);

  // 7) selective scan
  scan_kernel<<<DMODEL / 16, 256, 0, stream>>>(dt, xc, Bn, Cn, A_log, ys);

  // 8) y = (ys + xc*D) * silu(z) -> bf16
  gate_kernel<<<(L_SEQ * DMODEL) / 256, 256, 0, stream>>>(xzb, xc, D_param, ys, ysb);

  // 9) residual = hidden + y @ out_proj^T  [2048 x 1024 x 2048]
  gemm_wmma<2, false, false, false>
      <<<dim3(H_DIM / BN, L_SEQ / BM, 1), 256, 0, stream>>>(
          ysb, DMODEL, wb_out, nullptr, 0, DMODEL, residual, H_DIM, hidden, H_DIM,
          L_SEQ, H_DIM, DMODEL, nullptr, nullptr, nullptr, nullptr);

  // 10) x2 = rmsnorm(residual)
  rmsnorm_kernel<<<L_SEQ, 256, 0, stream>>>(residual, pre_ff_ln_w, xf, xnb, H_DIM);

  // 11) router softmax top-2 + token grouping
  router_kernel<<<L_SEQ, 256, 0, stream>>>(xf, router_w, top_idx, top_w);
  count_kernel<<<L_SEQ / 256, 256, 0, stream>>>(top_idx, counts);
  offsets_kernel<<<1, 32, 0, stream>>>(counts, offs);
  scatter_kernel<<<L_SEQ / 256, 256, 0, stream>>>(top_idx, top_w, offs, cursors,
                                                  rows_token, rows_weight);

  // 12) fused MoE gate+up: act = silu(x2 @ Wg^T) * (x2 @ Wu^T)  (gathered rows)
  gemm_wmma<5, true, true, true>
      <<<dim3(IDIM / BN, L_SEQ / BM, NEXP), 256, 0, stream>>>(
          xnb, H_DIM, wb_gate, wb_up, (long long)IDIM * H_DIM, H_DIM, actb, IDIM,
          nullptr, 0, L_SEQ, IDIM, H_DIM, counts, offs, rows_token, nullptr);

  // 13) out[token] += w * (act @ Wd^T)  [cnt x 1024 x 2048]
  gemm_wmma<3, true, false, false>
      <<<dim3(H_DIM / BN, L_SEQ / BM, NEXP), 256, 0, stream>>>(
          actb, IDIM, wb_down, nullptr, (long long)H_DIM * IDIM, IDIM, out, H_DIM,
          nullptr, 0, L_SEQ, H_DIM, IDIM, counts, offs, rows_token, rows_weight);
}
